// ESRNN_30717606101486
// MI455X (gfx1250) — matrix operand order repaired and checked by reading, not compiled, not run on hardware
//
#include <hip/hip_runtime.h>
#include <hip/hip_bf16.h>
#include <math.h>

typedef __attribute__((ext_vector_type(16))) _Float16 v16h;
typedef __attribute__((ext_vector_type(8)))  float    v8f;

#define NSER 1024
#define TT   256
#define SEASN 12
#define OUTW 12
#define HS   40
#define GATES 160
#define LWIN 245      /* W = T - IN + 1 */
#define LPRED 233     /* W - OUT */

__device__ __forceinline__ float sigmoidf_(float x) { return 1.0f / (1.0f + expf(-x)); }

/* ---------------- ES smoothing scan: one thread per series ---------------- */
__global__ void es_scan_kernel(const float* __restrict__ train,
                               const int*   __restrict__ idxv,
                               const float* __restrict__ lev_sms_p,
                               const float* __restrict__ seas_sms_p,
                               const float* __restrict__ init_seas,
                               float* __restrict__ seas,    /* [268][1024] */
                               float* __restrict__ levs,    /* [256][1024] */
                               float* __restrict__ lvlp)    /* [1024] */
{
    int b = blockIdx.x * blockDim.x + threadIdx.x;
    if (b >= NSER) return;
    int ib = idxv[b];
    float lsms = sigmoidf_(lev_sms_p[ib]);
    float ssms = sigmoidf_(seas_sms_p[ib]);
    float s0_0 = expf(init_seas[ib * SEASN + 0]);
    for (int j = 0; j < SEASN; ++j) seas[j * NSER + b] = expf(init_seas[ib * SEASN + j]);
    seas[SEASN * NSER + b] = s0_0;
    float lev = train[b * TT + 0] / s0_0;
    levs[0 * NSER + b] = lev;
    float prev_ld = 0.f, acc = 0.f;
    for (int t = 1; t < TT; ++t) {
        float s_t = seas[t * NSER + b];
        float x_t = train[b * TT + t];
        float new_lev = lsms * (x_t / s_t) + (1.f - lsms) * lev;
        float ld = logf(new_lev / lev);
        if (t >= 2) { float dd = ld - prev_ld; acc += dd * dd; }
        prev_ld = ld;
        float new_seas = ssms * (x_t / new_lev) + (1.f - ssms) * s_t;
        seas[(t + SEASN) * NSER + b] = new_seas;
        levs[t * NSER + b] = new_lev;
        lev = new_lev;
    }
    lvlp[b] = acc;
}

__global__ void loss_kernel(const float* __restrict__ part, float* __restrict__ out_loss)
{
    __shared__ float sh[256];
    int tid = threadIdx.x;
    float s = 0.f;
    for (int i = tid; i < NSER; i += 256) s += part[i];
    sh[tid] = s; __syncthreads();
    for (int off = 128; off > 0; off >>= 1) {
        if (tid < off) sh[tid] += sh[tid + off];
        __syncthreads();
    }
    if (tid == 0) out_loss[0] = sh[0] / (254.0f * (float)NSER);
}

/* ---------------- deseasonalize + window build ---------------- */
__global__ void deseas_kernel(const float* __restrict__ train,
                              const float* __restrict__ seas,
                              float* __restrict__ deseas)
{
    int i = blockIdx.x * 256 + threadIdx.x;   /* 256*1024 exact */
    int t = i >> 10, b = i & 1023;
    deseas[i] = train[b * TT + t] / seas[i];
}

__global__ void winin_kernel(const float* __restrict__ deseas,
                             const float* __restrict__ levs,
                             const float* __restrict__ info_cat,
                             float* __restrict__ X0)     /* [245][1024][18] */
{
    int i = blockIdx.x * 256 + threadIdx.x;   /* 245*1024*18 exact */
    int w = i / (NSER * 18);
    int r = i % (NSER * 18);
    int b = r / 18, j = r % 18;
    int anchor = 11 + w;
    if (j < 12) X0[i] = deseas[(w + j) * NSER + b] / levs[anchor * NSER + b];
    else        X0[i] = info_cat[b * 6 + (j - 12)];
}

__global__ void winout_kernel(const float* __restrict__ deseas,
                              const float* __restrict__ levs,
                              float* __restrict__ outB)   /* network_act [233][1024][12] */
{
    int i = blockIdx.x * 256 + threadIdx.x;   /* 233*1024*12 exact */
    int w = i / (NSER * OUTW);
    int r = i % (NSER * OUTW);
    int b = r / OUTW, j = r % OUTW;
    int anchor = 11 + w;
    outB[i] = deseas[(anchor + 1 + j) * NSER + b] / levs[anchor * NSER + b];
}

/* ---------------- fused dilated LSTM cell: WMMA core ----------------
 * F = input width (18 or 40), KC = #K-chunks of 32 (KPAD = 32*KC >= F+40).
 * Block = 64 threads = 2 waves; each wave owns 16 (phase,series) chains for
 * the full scan. Gates = [x | h] @ [Wih;Whh]^T + (bih+bhh) via
 * v_wmma_f32_16x16x32_f16 over 10 N-tiles x KC K-chunks.
 */
template<int F, int KC>
__global__ __launch_bounds__(64)
void lstm_cell_kernel(const float* __restrict__ X,   /* [245][1024][F]  */
                      float* __restrict__ Y,         /* [245][1024][40] */
                      const float* __restrict__ Wih, /* [160][F]  */
                      const float* __restrict__ Whh, /* [160][40] */
                      const float* __restrict__ bih,
                      const float* __restrict__ bhh,
                      const float* __restrict__ Res, /* residual or nullptr */
                      int d)
{
    constexpr int KPAD = KC * 32;
    constexpr int KTOT = F + HS;
    __shared__ _Float16 wfrag[10][KC][32][16];  /* B fragments, pre-swizzled */
    __shared__ float    bias_sh[GATES];
    __shared__ _Float16 a_sh[2][16][KPAD];      /* per-wave A staging [x|h] */
    __shared__ float    g_sh[2][16][GATES];     /* per-wave gate preacts    */

    const int tid  = threadIdx.x;
    const int wave = tid >> 5;
    const int lane = tid & 31;

    /* Build weight fragments in the exact WMMA B layout:
     * b[j] (j=0..15) = W_cat[N][k], k = 32*kc + (lane<16 ? 0:8) + (j<8 ? j : 8+j), N = 16*n + (lane&15) */
    for (int i = tid; i < 10 * KC * 32 * 16; i += 64) {
        int n  = i / (KC * 512);
        int r1 = i % (KC * 512);
        int kc = r1 / 512;
        int r2 = r1 % 512;
        int ln = r2 >> 4;
        int j  = r2 & 15;
        int k  = kc * 32 + ((ln < 16) ? 0 : 8) + ((j < 8) ? j : 8 + j);
        int col = n * 16 + (ln & 15);
        float w = 0.f;
        if (k < F)         w = Wih[col * F + k];
        else if (k < KTOT) w = Whh[col * HS + (k - F)];
        wfrag[n][kc][ln][j] = (_Float16)w;
    }
    for (int i = tid; i < GATES; i += 64) bias_sh[i] = bih[i] + bhh[i];
    for (int i = tid; i < 2 * 16 * KPAD; i += 64) ((_Float16*)a_sh)[i] = (_Float16)0.f;
    __syncthreads();

    const int rowBase = blockIdx.x * 32 + wave * 16;   /* row in [0, d*1024) */
    const int phase = rowBase >> 10;
    const int b0    = rowBase & 1023;
    const int nsteps = (LWIN + d - 1) / d;

    float creg[20];
    #pragma unroll
    for (int k = 0; k < 20; ++k) creg[k] = 0.f;

    const int Mrow  = lane & 15;
    const int loA   = (lane < 16) ? 0 : 8;
    const int Nl    = lane & 15;
    const int Mbase = (lane < 16) ? 0 : 8;

    for (int n = 0; n < nsteps; ++n) {
        const int t = n * d + phase;
        const bool valid = (t < LWIN);

        /* stage x rows (f32 -> f16) into a_sh[wave][r][0..F) */
        for (int i = lane; i < 16 * F; i += 32) {
            int r = i / F, j = i % F;
            float xv = valid ? X[((size_t)t * NSER + b0 + r) * F + j] : 0.f;
            a_sh[wave][r][j] = (_Float16)xv;
        }
        /* prefetch next step's activation tile while this step computes */
        if (t + d < LWIN) {
            const float* pf = X + ((size_t)(t + d) * NSER + b0) * F;
            __builtin_prefetch(pf + lane * 16, 0, 3);
            if (F > 32) __builtin_prefetch(pf + 512 + lane * 16, 0, 3);
        }
        __syncthreads();

        /* A fragments: 16x32 f16 layout */
        v16h afrag[KC];
        #pragma unroll
        for (int kc = 0; kc < KC; ++kc) {
            #pragma unroll
            for (int j = 0; j < 8; ++j) {
                afrag[kc][j]     = a_sh[wave][Mrow][kc * 32 + loA + j];
                afrag[kc][8 + j] = a_sh[wave][Mrow][kc * 32 + 16 + loA + j];
            }
        }

        #pragma unroll
        for (int nc = 0; nc < 10; ++nc) {
            float bv = bias_sh[nc * 16 + Nl];
            v8f acc;
            #pragma unroll
            for (int r = 0; r < 8; ++r) acc[r] = bv;
            #pragma unroll
            for (int kc = 0; kc < KC; ++kc) {
                v16h bf;
                #pragma unroll
                for (int j = 0; j < 16; ++j) bf[j] = wfrag[nc][kc][lane][j];
                acc = __builtin_amdgcn_wmma_f32_16x16x32_f16(
                          false, afrag[kc], false, bf, (short)0, acc, false, false);
            }
            #pragma unroll
            for (int r = 0; r < 8; ++r) g_sh[wave][Mbase + r][nc * 16 + Nl] = acc[r];
        }
        __syncthreads();

        /* gate nonlinearity: 640 (row,h) elems per wave, 20 per lane */
        #pragma unroll
        for (int k = 0; k < 20; ++k) {
            int e  = k * 32 + lane;
            int r  = e / HS;
            int hh = e % HS;
            float gi = g_sh[wave][r][hh];
            float gf = g_sh[wave][r][HS + hh];
            float gg = g_sh[wave][r][2 * HS + hh];
            float go = g_sh[wave][r][3 * HS + hh];
            float c = sigmoidf_(gf) * creg[k] + sigmoidf_(gi) * tanhf(gg);
            float h = sigmoidf_(go) * tanhf(c);
            creg[k] = c;
            a_sh[wave][r][F + hh] = (_Float16)h;
            if (valid) {
                size_t o = ((size_t)t * NSER + b0 + r) * (size_t)HS + hh;
                Y[o] = Res ? (h + Res[o]) : h;
            }
        }
        __syncthreads();
    }
}

/* ---------------- head: tanh(x@nlW^T+nlb) @ scW^T + scb, WMMA ----------------
 * Block = 64 threads = 2 waves; each wave does one 16-row tile.
 * nl: K=40 (2 chunks of 32), N=40 (3 tiles, cols>=40 padded/discarded)
 * sc: K=40 (2 chunks),       N=12 (1 tile,  cols>=12 discarded)          */
__global__ __launch_bounds__(64)
void head_wmma_kernel(const float* __restrict__ Yin,
                      const float* __restrict__ nlW, const float* __restrict__ nlb,
                      const float* __restrict__ scW, const float* __restrict__ scb,
                      float* __restrict__ out_pred, float* __restrict__ out_full)
{
    __shared__ _Float16 wfnl[3][2][32][16];
    __shared__ _Float16 wfsc[2][32][16];
    __shared__ float    bnl_sh[48];
    __shared__ float    bsc_sh[16];
    __shared__ _Float16 a_sh[2][16][64];
    __shared__ _Float16 h_sh[2][16][64];

    const int tid  = threadIdx.x;
    const int wave = tid >> 5;
    const int lane = tid & 31;

    for (int i = tid; i < 3 * 2 * 512; i += 64) {
        int n  = i / 1024;
        int r1 = i % 1024;
        int kc = r1 >> 9;
        int r2 = r1 & 511;
        int ln = r2 >> 4;
        int j  = r2 & 15;
        int k  = kc * 32 + ((ln < 16) ? 0 : 8) + ((j < 8) ? j : 8 + j);
        int col = n * 16 + (ln & 15);
        float w = (k < HS && col < HS) ? nlW[col * HS + k] : 0.f;
        wfnl[n][kc][ln][j] = (_Float16)w;
    }
    for (int i = tid; i < 2 * 512; i += 64) {
        int kc = i >> 9;
        int r2 = i & 511;
        int ln = r2 >> 4;
        int j  = r2 & 15;
        int k  = kc * 32 + ((ln < 16) ? 0 : 8) + ((j < 8) ? j : 8 + j);
        int col = ln & 15;
        float w = (k < HS && col < OUTW) ? scW[col * HS + k] : 0.f;
        wfsc[kc][ln][j] = (_Float16)w;
    }
    for (int i = tid; i < 48; i += 64) bnl_sh[i] = (i < HS) ? nlb[i] : 0.f;
    if (tid < 16) bsc_sh[tid] = (tid < OUTW) ? scb[tid] : 0.f;
    for (int i = tid; i < 2 * 16 * 64; i += 64) {
        ((_Float16*)a_sh)[i] = (_Float16)0.f;
        ((_Float16*)h_sh)[i] = (_Float16)0.f;
    }
    __syncthreads();

    const size_t rowBase = (size_t)blockIdx.x * 32 + (size_t)wave * 16;  /* 16-row tile */
    for (int i = lane; i < 16 * HS; i += 32) {
        int r = i / HS, j = i % HS;
        a_sh[wave][r][j] = (_Float16)Yin[(rowBase + r) * HS + j];
    }
    __syncthreads();

    const int Mrow  = lane & 15;
    const int loA   = (lane < 16) ? 0 : 8;
    const int Nl    = lane & 15;
    const int Mbase = (lane < 16) ? 0 : 8;

    v16h af[2];
    #pragma unroll
    for (int kc = 0; kc < 2; ++kc) {
        #pragma unroll
        for (int j = 0; j < 8; ++j) {
            af[kc][j]     = a_sh[wave][Mrow][kc * 32 + loA + j];
            af[kc][8 + j] = a_sh[wave][Mrow][kc * 32 + 16 + loA + j];
        }
    }

    #pragma unroll
    for (int nc = 0; nc < 3; ++nc) {
        float bv = bnl_sh[nc * 16 + Nl];
        v8f acc;
        #pragma unroll
        for (int r = 0; r < 8; ++r) acc[r] = bv;
        #pragma unroll
        for (int kc = 0; kc < 2; ++kc) {
            v16h bf;
            #pragma unroll
            for (int j = 0; j < 16; ++j) bf[j] = wfnl[nc][kc][lane][j];
            acc = __builtin_amdgcn_wmma_f32_16x16x32_f16(
                      false, af[kc], false, bf, (short)0, acc, false, false);
        }
        int col = nc * 16 + Nl;
        if (col < HS) {
            #pragma unroll
            for (int r = 0; r < 8; ++r)
                h_sh[wave][Mbase + r][col] = (_Float16)tanhf(acc[r]);
        }
    }
    __syncthreads();

    v16h af2[2];
    #pragma unroll
    for (int kc = 0; kc < 2; ++kc) {
        #pragma unroll
        for (int j = 0; j < 8; ++j) {
            af2[kc][j]     = h_sh[wave][Mrow][kc * 32 + loA + j];
            af2[kc][8 + j] = h_sh[wave][Mrow][kc * 32 + 16 + loA + j];
        }
    }
    {
        float bv = bsc_sh[Nl];
        v8f acc;
        #pragma unroll
        for (int r = 0; r < 8; ++r) acc[r] = bv;
        #pragma unroll
        for (int kc = 0; kc < 2; ++kc) {
            v16h bf;
            #pragma unroll
            for (int j = 0; j < 16; ++j) bf[j] = wfsc[kc][lane][j];
            acc = __builtin_amdgcn_wmma_f32_16x16x32_f16(
                      false, af2[kc], false, bf, (short)0, acc, false, false);
        }
        int t = (int)(rowBase >> 10);
        if (Nl < OUTW) {
            #pragma unroll
            for (int r = 0; r < 8; ++r) {
                size_t o = (rowBase + Mbase + r) * OUTW + Nl;
                out_full[o] = acc[r];
                if (t < LPRED) out_pred[o] = acc[r];   /* prefix == network_pred */
            }
        }
    }
}

/* ---------------- holdout / reseasonalize ---------------- */
__global__ void holdout_kernel(const float* __restrict__ out_full,
                               const float* __restrict__ seas,
                               const float* __restrict__ levs,
                               const float* __restrict__ valp,
                               float* __restrict__ out_hop,
                               float* __restrict__ out_hoa,
                               float* __restrict__ out_hod)
{
    int i = blockIdx.x * blockDim.x + threadIdx.x;
    if (i >= NSER * OUTW) return;
    int b = i / OUTW, j = i % OUTW;
    float sfin = seas[(TT + j) * NSER + b];           /* seas_stacked[:, -12:] */
    float lev  = levs[255 * NSER + b];
    float r = out_full[((size_t)(LWIN - 1) * NSER + b) * OUTW + j] * sfin * lev;
    out_hop[i] = (r > 0.f) ? r : 0.f;
    float v = valp[i];
    out_hoa[i] = v;
    out_hod[i] = v / sfin / lev;
}

extern "C" void kernel_launch(void* const* d_in, const int* in_sizes, int n_in,
                              void* d_out, int out_size, void* d_ws, size_t ws_size,
                              hipStream_t stream)
{
    (void)out_size; (void)ws_size;
    const float *train, *valp, *test, *info_cat, *lev_sms, *seas_sms, *init_seas;
    const int* idxv;
    const float *Wih[4], *Whh[4], *bih[4], *bhh[4], *nlW, *nlb, *scW, *scb;
    auto Fp = [&](int i) { return (const float*)d_in[i]; };

    if (n_in >= 28 && in_sizes[0] == NSER * TT) {
        /* top-level dict order: train, val, test, info_cat, idx, params... */
        train = Fp(0); valp = Fp(1); test = Fp(2); info_cat = Fp(3);
        idxv = (const int*)d_in[4];
        if (in_sizes[5] == NSER) {
            /* params in insertion order */
            lev_sms = Fp(5); seas_sms = Fp(6); init_seas = Fp(7);
            for (int c = 0; c < 4; ++c) {
                Wih[c] = Fp(8 + 4 * c); Whh[c] = Fp(9 + 4 * c);
                bih[c] = Fp(10 + 4 * c); bhh[c] = Fp(11 + 4 * c);
            }
        } else {
            /* params pytree-flattened (alphabetical): drnn first, cell dict Whh,Wih,bhh,bih */
            for (int c = 0; c < 4; ++c) {
                Whh[c] = Fp(5 + 4 * c); Wih[c] = Fp(6 + 4 * c);
                bhh[c] = Fp(7 + 4 * c); bih[c] = Fp(8 + 4 * c);
            }
            lev_sms = Fp(21); seas_sms = Fp(22); init_seas = Fp(23);
        }
        nlW = Fp(24); nlb = Fp(25); scW = Fp(26); scb = Fp(27);
    } else {
        /* fully alphabetical pytree flatten of setup_inputs() */
        idxv = (const int*)d_in[0]; info_cat = Fp(1);
        for (int c = 0; c < 4; ++c) {
            Whh[c] = Fp(2 + 4 * c); Wih[c] = Fp(3 + 4 * c);
            bhh[c] = Fp(4 + 4 * c); bih[c] = Fp(5 + 4 * c);
        }
        lev_sms = Fp(18); seas_sms = Fp(19); init_seas = Fp(20);
        nlW = Fp(21); nlb = Fp(22); scW = Fp(23); scb = Fp(24);
        test = Fp(25); train = Fp(26); valp = Fp(27);
    }
    (void)test;

    float* ws = (float*)d_ws;
    float* seas   = ws;                              /* 268*1024 */
    float* levs   = seas + 268 * NSER;               /* 256*1024 */
    float* deseas = levs + 256 * NSER;               /* 256*1024 */
    float* lvlp   = deseas + 256 * NSER;             /* 1024 */
    float* BUF0   = lvlp + NSER;                     /* 245*1024*40 (X0 w/ stride 18 first) */
    float* BUF1   = BUF0 + (size_t)LWIN * NSER * HS;
    float* BUF2   = BUF1 + (size_t)LWIN * NSER * HS;

    float* out    = (float*)d_out;
    float* o_pred = out;                                    /* 233*1024*12 */
    float* o_act  = o_pred + (size_t)LPRED * NSER * OUTW;   /* 233*1024*12 */
    float* o_hop  = o_act  + (size_t)LPRED * NSER * OUTW;   /* 1024*12 */
    float* o_full = o_hop  + (size_t)NSER * OUTW;           /* 245*1024*12 */
    float* o_hoa  = o_full + (size_t)LWIN * NSER * OUTW;    /* 1024*12 */
    float* o_hod  = o_hoa  + (size_t)NSER * OUTW;           /* 1024*12 */
    float* o_loss = o_hod  + (size_t)NSER * OUTW;           /* 1 */

    es_scan_kernel<<<8, 128, 0, stream>>>(train, idxv, lev_sms, seas_sms, init_seas,
                                          seas, levs, lvlp);
    loss_kernel<<<1, 256, 0, stream>>>(lvlp, o_loss);
    deseas_kernel<<<1024, 256, 0, stream>>>(train, seas, deseas);
    winin_kernel<<<17640, 256, 0, stream>>>(deseas, levs, info_cat, BUF0);
    winout_kernel<<<11184, 256, 0, stream>>>(deseas, levs, o_act);

    /* drnn stack: group0 (d=1,2), group1 (d=4,8) with residual into last cell */
    lstm_cell_kernel<18, 2><<<32,  64, 0, stream>>>(BUF0, BUF1, Wih[0], Whh[0], bih[0], bhh[0], nullptr, 1);
    lstm_cell_kernel<40, 3><<<64,  64, 0, stream>>>(BUF1, BUF2, Wih[1], Whh[1], bih[1], bhh[1], nullptr, 2);
    lstm_cell_kernel<40, 3><<<128, 64, 0, stream>>>(BUF2, BUF0, Wih[2], Whh[2], bih[2], bhh[2], nullptr, 4);
    lstm_cell_kernel<40, 3><<<256, 64, 0, stream>>>(BUF0, BUF1, Wih[3], Whh[3], bih[3], bhh[3], BUF2, 8);

    head_wmma_kernel<<<7840, 64, 0, stream>>>(BUF1, nlW, nlb, scW, scb, o_pred, o_full);
    holdout_kernel<<<48, 256, 0, stream>>>(o_full, seas, levs, valp, o_hop, o_hoa, o_hod);
}
